// Net_63256278336098
// MI455X (gfx1250) — matrix operand order repaired and checked by reading, not compiled, not run on hardware
//
#include <hip/hip_runtime.h>
#include <hip/hip_bf16.h>

// ---------------------------------------------------------------------------
// GIN (2x GINConv(128->128->128) + global_add_pool + MLP head) for MI455X.
//   - agg buffer initialized with x itself (GIN self-term), then scatter-add
//     of neighbors with f32 atomics (L2-resident: 51MB < 192MB L2)
//   - fused per-conv kernel: agg -> bf16 LDS (packed b64 stores) -> WMMA GEMM1
//     -> relu -> LDS (in place, wave-private rows => no barriers) -> WMMA
//     GEMM2 -> relu -> f32 out. 64x v_wmma_f32_16x16x32_bf16 per wave.
//   - weights pre-converted to bf16, transposed [out][in] so B fragments are
//     contiguous 32B runs -> compiler emits global_load_b128 pairs.
// ---------------------------------------------------------------------------

#define N_NODES    100000
#define N_EDGES    1600000
#define NUM_GRAPHS 512
#define DIM        128
#define LDS_STRIDE 136   // 128 + 8 halfword pad: breaks 256B bank period

typedef __attribute__((ext_vector_type(16))) __bf16 v16bf;
typedef __attribute__((ext_vector_type(8)))  float  v8f;

union FragBF { unsigned int u[8]; v16bf v; };

__device__ __forceinline__ unsigned int f2bf(float f) {
    unsigned int u = __float_as_uint(f);
    u = (u + 0x7FFFu + ((u >> 16) & 1u)) >> 16;   // round-to-nearest-even
    return u;
}

// ---------------------------------------------------------------------------
// Convert the four conv-MLP weight matrices to bf16, transposed [out][in].
// w layout from jax: [fan_in][fan_out] row-major.
__global__ __launch_bounds__(256) void prep_w_kernel(
    const float* __restrict__ w0, const float* __restrict__ w1,
    const float* __restrict__ w2, const float* __restrict__ w3,
    unsigned short* __restrict__ wt)
{
    int tid = blockIdx.x * 256 + threadIdx.x;
    if (tid >= 4 * DIM * DIM) return;
    int wsel = tid >> 14;
    int rem  = tid & 16383;
    int o = rem >> 7;        // out channel
    int i = rem & 127;       // in channel
    const float* w = (wsel == 0) ? w0 : (wsel == 1) ? w1 : (wsel == 2) ? w2 : w3;
    wt[(size_t)wsel * DIM * DIM + o * DIM + i] =
        (unsigned short)f2bf(w[i * DIM + o]);
}

__global__ __launch_bounds__(256) void zero_kernel(float* __restrict__ p, int n)
{
    int tid = blockIdx.x * 256 + threadIdx.x;
    if (tid < n) p[tid] = 0.0f;
}

// dst = src  (float4-wide): initializes agg with the GIN self-term.
__global__ __launch_bounds__(256) void copy_kernel(
    const float* __restrict__ src, float* __restrict__ dst, int n4)
{
    int tid = blockIdx.x * 256 + threadIdx.x;
    if (tid < n4)
        ((float4*)dst)[tid] = ((const float4*)src)[tid];
}

// ---------------------------------------------------------------------------
// agg[dst] += x[src]   : 32 threads per edge, float4 per thread, f32 atomics.
__global__ __launch_bounds__(256) void scatter_kernel(
    const float* __restrict__ x, const int* __restrict__ src,
    const int* __restrict__ dst, float* __restrict__ agg)
{
    int tid = blockIdx.x * 256 + threadIdx.x;
    int e   = tid >> 5;
    if (e >= N_EDGES) return;
    int c4 = (tid & 31) << 2;
    int s = src[e], d = dst[e];
    const float4 v = *(const float4*)(x + (size_t)s * DIM + c4);
    float* p = agg + (size_t)d * DIM + c4;
    atomicAdd(p + 0, v.x);
    atomicAdd(p + 1, v.y);
    atomicAdd(p + 2, v.z);
    atomicAdd(p + 3, v.w);
}

// ---------------------------------------------------------------------------
// Fused GIN MLP: hout = relu(relu(hin@W1 + b1)@W2 + b2)
// hin already holds x + sum_neighbors(x).
// Block = 256 threads = 8 waves; each wave owns 16 node-rows (wave-private ->
// no __syncthreads needed).
__global__ __launch_bounds__(256) void gin_mlp_kernel(
    const float* __restrict__ hin,
    const unsigned short* __restrict__ w1t, const float* __restrict__ b1,
    const unsigned short* __restrict__ w2t, const float* __restrict__ b2,
    float* __restrict__ hout)
{
    __shared__ unsigned short sA[128 * LDS_STRIDE];   // 34,816 B

    const int wave = threadIdx.x >> 5;
    const int lane = threadIdx.x & 31;
    const int r0   = wave * 16;                 // this wave's first row in tile
    const int base = blockIdx.x * 128;          // first node of this block

    // ---- stage hin as bf16 into wave-private LDS rows (one b64 store/row) ----
    #pragma unroll 4
    for (int r = 0; r < 16; ++r) {
        int node = base + r0 + r;
        float4 v = make_float4(0.f, 0.f, 0.f, 0.f);
        if (node < N_NODES)
            v = *(const float4*)(hin + (size_t)node * DIM + lane * 4);
        uint2 pk;
        pk.x = f2bf(v.x) | (f2bf(v.y) << 16);
        pk.y = f2bf(v.z) | (f2bf(v.w) << 16);
        *(uint2*)&sA[(r0 + r) * LDS_STRIDE + lane * 4] = pk;
    }
    // same-wave LDS RAW handled by compiler-inserted s_wait_dscnt

    const int m   = lane & 15;       // row within 16x16 tile (A / C-D mapping)
    const int n   = lane & 15;       // col within tile (B / C-D mapping)
    const int sel = lane >> 4;       // lane-half select

    const v8f vzero = {0.f, 0.f, 0.f, 0.f, 0.f, 0.f, 0.f, 0.f};

    // ================= GEMM 1 : hin @ W1, +b1, relu =================
    v8f acc[8];
    #pragma unroll
    for (int nt = 0; nt < 8; ++nt) acc[nt] = vzero;

    #pragma unroll
    for (int s = 0; s < 4; ++s) {                 // K steps of 32
        FragBF a;
        #pragma unroll
        for (int j = 0; j < 8; ++j) {
            // A k order chosen here (same as B): VGPR j -> K = sel*16 + 2j
            int kk = s * 32 + (sel << 4) + (j << 1);
            a.u[j] = *(const unsigned int*)&sA[(r0 + m) * LDS_STRIDE + kk];
        }
        #pragma unroll
        for (int nt = 0; nt < 8; ++nt) {
            FragBF bm;
            #pragma unroll
            for (int j = 0; j < 8; ++j) {
                // B k order chosen here (same as A): lanes 0-15 K=0..15, 16-31 K=16..31
                int kk = s * 32 + (sel << 4) + (j << 1);
                bm.u[j] = *(const unsigned int*)&w1t[((nt * 16 + n) << 7) + kk];
            }
            acc[nt] = __builtin_amdgcn_wmma_f32_16x16x32_bf16(
                false, a.v, false, bm.v, (short)0, acc[nt], false, false);
        }
    }

    // bias + relu, overwrite own LDS rows with bf16 h1
    #pragma unroll
    for (int nt = 0; nt < 8; ++nt) {
        float bv = b1[nt * 16 + n];
        #pragma unroll
        for (int r = 0; r < 8; ++r) {
            float t = acc[nt][r] + bv;
            t = t > 0.f ? t : 0.f;
            sA[(r0 + r + sel * 8) * LDS_STRIDE + nt * 16 + n] =
                (unsigned short)f2bf(t);
        }
    }

    // ================= GEMM 2 : h1 @ W2, +b2, relu, store f32 =============
    #pragma unroll
    for (int nt = 0; nt < 8; ++nt) acc[nt] = vzero;

    #pragma unroll
    for (int s = 0; s < 4; ++s) {
        FragBF a;
        #pragma unroll
        for (int j = 0; j < 8; ++j) {
            int kk = s * 32 + (sel << 4) + (j << 1);
            a.u[j] = *(const unsigned int*)&sA[(r0 + m) * LDS_STRIDE + kk];
        }
        #pragma unroll
        for (int nt = 0; nt < 8; ++nt) {
            FragBF bm;
            #pragma unroll
            for (int j = 0; j < 8; ++j) {
                int kk = s * 32 + (sel << 4) + (j << 1);
                bm.u[j] = *(const unsigned int*)&w2t[((nt * 16 + n) << 7) + kk];
            }
            acc[nt] = __builtin_amdgcn_wmma_f32_16x16x32_bf16(
                false, a.v, false, bm.v, (short)0, acc[nt], false, false);
        }
    }

    #pragma unroll
    for (int nt = 0; nt < 8; ++nt) {
        float bv = b2[nt * 16 + n];
        #pragma unroll
        for (int r = 0; r < 8; ++r) {
            int node = base + r0 + r + sel * 8;
            float t = acc[nt][r] + bv;
            t = t > 0.f ? t : 0.f;
            if (node < N_NODES)
                hout[(size_t)node * DIM + nt * 16 + n] = t;
        }
    }
}

// ---------------------------------------------------------------------------
// g[batch[node]] += h[node]
__global__ __launch_bounds__(256) void pool_kernel(
    const float* __restrict__ h, const int* __restrict__ batch,
    float* __restrict__ g)
{
    int tid  = blockIdx.x * 256 + threadIdx.x;
    int node = tid >> 5;
    if (node >= N_NODES) return;
    int c4 = (tid & 31) << 2;
    int gi = batch[node];
    const float4 v = *(const float4*)(h + (size_t)node * DIM + c4);
    float* p = g + (size_t)gi * DIM + c4;
    atomicAdd(p + 0, v.x);
    atomicAdd(p + 1, v.y);
    atomicAdd(p + 2, v.z);
    atomicAdd(p + 3, v.w);
}

// ---------------------------------------------------------------------------
// out[g] = relu(g_row @ lin1 + b1) @ lin2 + b2   (512 graphs, tiny)
__global__ __launch_bounds__(128) void head_kernel(
    const float* __restrict__ g, const float* __restrict__ w1,
    const float* __restrict__ b1, const float* __restrict__ w2,
    const float* __restrict__ b2, float* __restrict__ out)
{
    __shared__ float row[128];
    __shared__ float red[128];
    const int t = threadIdx.x;
    const int gr = blockIdx.x;
    row[t] = g[(size_t)gr * DIM + t];
    __syncthreads();
    float s = b1[t];
    #pragma unroll 8
    for (int k = 0; k < 128; ++k) s += row[k] * w1[k * DIM + t];
    s = s > 0.f ? s : 0.f;
    red[t] = s * w2[t];
    __syncthreads();
    #pragma unroll
    for (int off = 64; off > 0; off >>= 1) {
        if (t < off) red[t] += red[t + off];
        __syncthreads();
    }
    if (t == 0) out[gr] = red[0] + b2[0];
}

// ---------------------------------------------------------------------------
extern "C" void kernel_launch(void* const* d_in, const int* in_sizes, int n_in,
                              void* d_out, int out_size, void* d_ws, size_t ws_size,
                              hipStream_t stream)
{
    const float* x      = (const float*)d_in[0];
    const int*   ei     = (const int*)  d_in[1];   // [2, N_EDGES]
    const int*   batch  = (const int*)  d_in[2];
    const float* c1w1   = (const float*)d_in[3];
    const float* c1b1   = (const float*)d_in[4];
    const float* c1w2   = (const float*)d_in[5];
    const float* c1b2   = (const float*)d_in[6];
    const float* c2w1   = (const float*)d_in[7];
    const float* c2b1   = (const float*)d_in[8];
    const float* c2w2   = (const float*)d_in[9];
    const float* c2b2   = (const float*)d_in[10];
    const float* lin1_w = (const float*)d_in[11];
    const float* lin1_b = (const float*)d_in[12];
    const float* lin2_w = (const float*)d_in[13];
    const float* lin2_b = (const float*)d_in[14];
    float* out = (float*)d_out;

    const int* src = ei;
    const int* dst = ei + N_EDGES;

    // workspace layout (all offsets 256B aligned)
    const size_t NB = (size_t)N_NODES * DIM * sizeof(float);     // 51,200,000
    char* ws = (char*)d_ws;
    float*          agg   = (float*)(ws);
    float*          hbf   = (float*)(ws + NB);
    float*          gpool = (float*)(ws + 2 * NB);               // 512*128 f32
    unsigned short* wt    = (unsigned short*)(ws + 2 * NB +
                                (size_t)NUM_GRAPHS * DIM * sizeof(float));

    const int nodeBlocks  = (N_NODES + 127) / 128;               // 782
    const int scatBlocks  = (N_EDGES * 32) / 256;                // 200,000
    const int copyBlocks  = (N_NODES * DIM / 4 + 255) / 256;     // float4 copy
    const int gFloats     = NUM_GRAPHS * DIM;
    const int zeroBlocksG = (gFloats + 255) / 256;
    const int poolBlocks  = (N_NODES * 32 + 255) / 256;

    // 0) bf16-transpose all conv weights
    prep_w_kernel<<<(4 * DIM * DIM + 255) / 256, 256, 0, stream>>>(
        c1w1, c1w2, c2w1, c2w2, wt);

    // ---- conv 1 : agg = x + sum_neighbors(x); MLP -> hbf ----
    copy_kernel<<<copyBlocks, 256, 0, stream>>>(x, agg, N_NODES * DIM / 4);
    scatter_kernel<<<scatBlocks, 256, 0, stream>>>(x, src, dst, agg);
    gin_mlp_kernel<<<nodeBlocks, 256, 0, stream>>>(
        agg, wt + 0 * DIM * DIM, c1b1, wt + 1 * DIM * DIM, c1b2, hbf);

    // ---- conv 2 : agg = h + sum_neighbors(h); MLP -> hbf (disjoint rows) ----
    copy_kernel<<<copyBlocks, 256, 0, stream>>>(hbf, agg, N_NODES * DIM / 4);
    scatter_kernel<<<scatBlocks, 256, 0, stream>>>(hbf, src, dst, agg);
    gin_mlp_kernel<<<nodeBlocks, 256, 0, stream>>>(
        agg, wt + 2 * DIM * DIM, c2b1, wt + 3 * DIM * DIM, c2b2, hbf);

    // ---- global add pool + head ----
    zero_kernel<<<zeroBlocksG, 256, 0, stream>>>(gpool, gFloats);
    pool_kernel<<<poolBlocks, 256, 0, stream>>>(hbf, batch, gpool);
    head_kernel<<<NUM_GRAPHS, 128, 0, stream>>>(
        gpool, lin1_w, lin1_b, lin2_w, lin2_b, out);
}
